// Block_77524159693337
// MI455X (gfx1250) — matrix operand order repaired and checked by reading, not compiled, not run on hardware
//
#include <hip/hip_runtime.h>
#include <hip/hip_bf16.h>
#include <stdint.h>

// ---------------------------------------------------------------------------
// Transformer block for MI455X (gfx1250): LN -> MHA(causal) -> +res -> LN ->
// MLP(ReLU) -> +res.  All matmuls on v_wmma_f32_16x16x32_bf16 (f32 accum).
// GEMMs use double-buffered LDS tiles, filled via async global->LDS copies
// (GLOBAL_LOAD_ASYNC_TO_LDS_B128 / ASYNCcnt) when the toolchain exposes them.
// ---------------------------------------------------------------------------

static constexpr int kB  = 2;
static constexpr int kT  = 2048;
static constexpr int kC  = 1024;
static constexpr int kNH = 16;
static constexpr int kH  = 64;
static constexpr int kFF = 3 * kC;      // 3072
static constexpr int kM  = kB * kT;     // 4096 rows
static constexpr float kEPS = 1e-6f;

typedef __attribute__((ext_vector_type(16))) __bf16        bf16x16;
typedef __attribute__((ext_vector_type(8)))  float         f32x8;
typedef __attribute__((ext_vector_type(4)))  unsigned int  u32x4;
typedef __attribute__((ext_vector_type(4)))  int           i32x4;

union FragB16 {
    bf16x16 v;
    u32x4   q[2];
    unsigned short s[16];
};

// ---- async global->LDS path (probe-safe) ----------------------------------
#if __has_builtin(__builtin_amdgcn_global_load_async_to_lds_b128) && \
    __has_builtin(__builtin_amdgcn_s_wait_asynccnt)
#define USE_ASYNC_LDS 1
#else
#define USE_ASYNC_LDS 0
#endif

__device__ __forceinline__ void copy16_to_lds(const unsigned short* g,
                                              unsigned short* l) {
#if USE_ASYNC_LDS
    __builtin_amdgcn_global_load_async_to_lds_b128(
        (__attribute__((address_space(1))) i32x4*)g,
        (__attribute__((address_space(3))) i32x4*)l,
        /*offset=*/0, /*cpol=*/0);
#else
    *(u32x4*)l = *(const u32x4*)g;
#endif
}

__device__ __forceinline__ void async_copy_wait() {
#if USE_ASYNC_LDS
    __builtin_amdgcn_s_wait_asynccnt(0);
#endif
}

__device__ __forceinline__ unsigned short f2bf(float f) {
    union { float f; unsigned int u; } c; c.f = f;
    unsigned int u = c.u;
    unsigned int r = u + 0x7fffu + ((u >> 16) & 1u);   // round-to-nearest-even
    return (unsigned short)(r >> 16);
}

__device__ __forceinline__ f32x8 wmma_bf16(const FragB16& a, const FragB16& b, f32x8 c) {
    return __builtin_amdgcn_wmma_f32_16x16x32_bf16(
        /*neg_a=*/false, a.v, /*neg_b=*/false, b.v,
        /*c_mod=*/(short)0, c, /*reuse_a=*/false, /*reuse_b=*/false);
}

// ---------------------------------------------------------------------------
// Weight prep: Wq/Wk/Wv [NH,C,H] f32  ->  Wqkv_t [3072 rows=out][1024=in] bf16
// ---------------------------------------------------------------------------
__global__ __launch_bounds__(256) void k_prep_qkv(
    const float* __restrict__ Wq, const float* __restrict__ Wk,
    const float* __restrict__ Wv, unsigned short* __restrict__ out)
{
    int idx = blockIdx.x * 256 + threadIdx.x;       // 0 .. 3*1024*1024-1
    int n = idx >> 10;                               // out row 0..3071
    int c = idx & 1023;                              // in col
    const float* src = (n < kC) ? Wq : (n < 2 * kC ? Wk : Wv);
    int nn = n & (kC - 1);
    int h = nn >> 6, j = nn & (kH - 1);
    float v = src[((size_t)h * kC + c) * kH + j];
    out[idx] = f2bf(v);
}

__global__ __launch_bounds__(256) void k_cvt_bf16(
    const float* __restrict__ in, unsigned short* __restrict__ out, int n)
{
    int idx = blockIdx.x * 256 + threadIdx.x;
    if (idx < n) out[idx] = f2bf(in[idx]);
}

// ---------------------------------------------------------------------------
// LayerNorm (torch semantics: unbiased std, eps added to std), bf16 output
// One block of 256 threads per row (C = 1024 -> 4 elems/thread).
// ---------------------------------------------------------------------------
__global__ __launch_bounds__(256) void k_layernorm(
    const float* __restrict__ x, const float* __restrict__ gamma,
    const float* __restrict__ beta, unsigned short* __restrict__ out)
{
    const int row = blockIdx.x;
    const float* xr = x + (size_t)row * kC;
    float lx[4];
    float s = 0.f, ss = 0.f;
    #pragma unroll
    for (int i = 0; i < 4; i++) {
        float v = xr[threadIdx.x + i * 256];
        lx[i] = v; s += v; ss += v * v;
    }
    #pragma unroll
    for (int m = 16; m >= 1; m >>= 1) {
        s  += __shfl_xor(s,  m, 32);
        ss += __shfl_xor(ss, m, 32);
    }
    __shared__ float ps[8], pss[8];
    if ((threadIdx.x & 31) == 0) { ps[threadIdx.x >> 5] = s; pss[threadIdx.x >> 5] = ss; }
    __syncthreads();
    s = 0.f; ss = 0.f;
    #pragma unroll
    for (int w = 0; w < 8; w++) { s += ps[w]; ss += pss[w]; }
    float mean = s * (1.0f / kC);
    float var  = (ss - (float)kC * mean * mean) * (1.0f / (kC - 1));
    float inv  = 1.0f / (sqrtf(fmaxf(var, 0.f)) + kEPS);
    #pragma unroll
    for (int i = 0; i < 4; i++) {
        int c = threadIdx.x + i * 256;
        float y = gamma[c] * (lx[i] - mean) * inv + beta[c];
        out[(size_t)row * kC + c] = f2bf(y);
    }
}

// ---------------------------------------------------------------------------
// Generic bf16 GEMM:  out[M,N] = act( A[M,K] * W[N,K]^T + bias + resid )
// Block tile 128x128, K-tile 32, 256 threads = 8 waves (4 in M x 2 in N),
// per-wave 2x4 subtiles of 16x16 -> 8 v_wmma per K-step.
// Double-buffered LDS; tiles filled by async global->LDS copies when available.
// ---------------------------------------------------------------------------
static constexpr int GBM = 128, GBN = 128, GBK = 32;
static constexpr int GLS = GBK + 8;     // LDS row stride (ushorts), bank-spread

template<bool BIAS, bool RELU, bool RESID, bool OUTBF16>
__global__ __launch_bounds__(256) void k_gemm(
    const unsigned short* __restrict__ A,   // [M,K] bf16
    const unsigned short* __restrict__ W,   // [N,K] bf16 (row = output feature)
    const float* __restrict__ bias,         // [N] or nullptr
    const float* __restrict__ resid,        // [M,N] or nullptr
    void* __restrict__ outp,                // bf16 or f32 [M,N]
    int M, int N, int K)
{
    __shared__ unsigned short As[2][GBM * GLS];   // 2 x 10 KB
    __shared__ unsigned short Ws[2][GBN * GLS];   // 2 x 10 KB

    const int tid  = threadIdx.x;
    const int lane = tid & 31;
    const int wave = tid >> 5;
    const int wm   = wave & 3;          // wave row (4)
    const int wn   = wave >> 2;         // wave col (2)
    const int bm0  = blockIdx.y * GBM;
    const int bn0  = blockIdx.x * GBN;
    const int l15  = lane & 15;
    const int half = lane >> 4;

    f32x8 acc[2][4];
    #pragma unroll
    for (int i = 0; i < 2; i++)
        #pragma unroll
        for (int j = 0; j < 4; j++)
            #pragma unroll
            for (int g = 0; g < 8; g++) acc[i][j][g] = 0.f;

    const int lr = tid >> 1;            // 0..127 tile row loaded by this thread
    const int lc = (tid & 1) * 16;      // 0 or 16 (16 bf16 = 32B each)

    auto load_tile = [&](int k0, int buf) {
        const unsigned short* sa = A + (size_t)(bm0 + lr) * K + k0 + lc;
        const unsigned short* sw = W + (size_t)(bn0 + lr) * K + k0 + lc;
        unsigned short* da = &As[buf][lr * GLS + lc];
        unsigned short* dw = &Ws[buf][lr * GLS + lc];
        copy16_to_lds(sa,     da);
        copy16_to_lds(sa + 8, da + 8);
        copy16_to_lds(sw,     dw);
        copy16_to_lds(sw + 8, dw + 8);
    };

    const int nk = K / GBK;
    load_tile(0, 0);
    async_copy_wait();
    __syncthreads();

    for (int kt = 0; kt < nk; kt++) {
        const int buf = kt & 1;
        if (kt + 1 < nk) load_tile((kt + 1) * GBK, buf ^ 1);

        FragB16 af[2], wf[4];
        #pragma unroll
        for (int i = 0; i < 2; i++) {
            // A frag (16x32 bf16): lane row = l15; K chunks at half*8 and +16
            const unsigned short* p = &As[buf][(wm * 32 + i * 16 + l15) * GLS + half * 8];
            af[i].q[0] = *(const u32x4*)p;
            af[i].q[1] = *(const u32x4*)(p + 16);
        }
        #pragma unroll
        for (int j = 0; j < 4; j++) {
            // B frag (32x16): lane col = l15; K chunks at half*8 and +16
            const unsigned short* p = &Ws[buf][(wn * 64 + j * 16 + l15) * GLS + half * 8];
            wf[j].q[0] = *(const u32x4*)p;
            wf[j].q[1] = *(const u32x4*)(p + 16);
        }
        #pragma unroll
        for (int i = 0; i < 2; i++)
            #pragma unroll
            for (int j = 0; j < 4; j++)
                acc[i][j] = wmma_bf16(af[i], wf[j], acc[i][j]);

        if (kt + 1 < nk) {
            async_copy_wait();
            __syncthreads();
        }
    }

    // Epilogue.  C/D layout: VGPR g -> row g + 8*half, col = l15.
    #pragma unroll
    for (int j = 0; j < 4; j++) {
        const int ng = bn0 + wn * 64 + j * 16 + l15;
        const float bv = BIAS ? bias[ng] : 0.f;
        #pragma unroll
        for (int i = 0; i < 2; i++) {
            #pragma unroll
            for (int g = 0; g < 8; g++) {
                const int mg = bm0 + wm * 32 + i * 16 + half * 8 + g;
                float v = acc[i][j][g] + bv;
                if (RESID) v += resid[(size_t)mg * N + ng];
                if (RELU)  v = fmaxf(v, 0.f);
                if (OUTBF16) ((unsigned short*)outp)[(size_t)mg * N + ng] = f2bf(v);
                else         ((float*)outp)[(size_t)mg * N + ng] = v;
            }
        }
    }
}

// ---------------------------------------------------------------------------
// V transpose: qkv[:, 2048 + h*64 + j]  ->  vt[(b*16+h)*64 + j][t]  (bf16)
// so attention PV B-fragments are two contiguous 16B loads per lane.
// ---------------------------------------------------------------------------
__global__ __launch_bounds__(256) void k_transpose_v(
    const unsigned short* __restrict__ qkv, unsigned short* __restrict__ vt)
{
    __shared__ unsigned short tile[32][72];
    const int bh = blockIdx.y;               // 0..31
    const int s0 = blockIdx.x * 32;
    const int b = bh >> 4, h = bh & 15;
    const int tid = threadIdx.x;
    {
        int r  = tid >> 3;                   // 0..31 (s within tile)
        int j0 = (tid & 7) * 8;              // 0..56
        const unsigned short* src =
            qkv + (size_t)(b * kT + s0 + r) * (3 * kC) + 2 * kC + h * kH + j0;
        *(u32x4*)&tile[r][j0] = *(const u32x4*)src;
    }
    __syncthreads();
    {
        int j  = tid >> 2;                   // 0..63
        int sb = (tid & 3) * 8;              // 0,8,16,24
        union { unsigned short s[8]; u32x4 q; } tmp;
        #pragma unroll
        for (int k = 0; k < 8; k++) tmp.s[k] = tile[sb + k][j];
        unsigned short* dst = vt + ((size_t)bh * kH + j) * kT + s0 + sb;
        *(u32x4*)dst = tmp.q;
    }
}

// ---------------------------------------------------------------------------
// Causal flash attention.  Grid: (T/128, B*NH); 8 waves, 16 q-rows per wave.
// Scores: Q(16x64) x K^T via 2x v_wmma per 16-col subtile; online softmax with
// 16-lane-half shfl_xor reductions; P repacked C-layout -> A-layout via LDS.
// ---------------------------------------------------------------------------
__global__ __launch_bounds__(256) void k_attention(
    const unsigned short* __restrict__ qkv,   // [M, 3072] bf16 (Q|K|V)
    const unsigned short* __restrict__ vt,    // [B*NH*H, T] bf16 (V^T)
    unsigned short* __restrict__ ctx)         // [M, 1024] bf16
{
    __shared__ unsigned short pbuf[8][16 * 40];   // per-wave P scratch
    const int lane = threadIdx.x & 31;
    const int wave = threadIdx.x >> 5;
    const int l15  = lane & 15;
    const int half = lane >> 4;
    const int bh   = blockIdx.y;
    const int b    = bh >> 4, h = bh & 15;
    const int t0   = blockIdx.x * 128 + wave * 16;

    // Q fragments: A-layout, lane row = l15, chunks over H=64 (2 x K32)
    FragB16 aq[2];
    {
        const unsigned short* qb =
            qkv + (size_t)(b * kT + t0 + l15) * (3 * kC) + h * kH;
        #pragma unroll
        for (int cfr = 0; cfr < 2; cfr++) {
            int ko = cfr * 32 + half * 8;
            aq[cfr].q[0] = *(const u32x4*)(qb + ko);
            aq[cfr].q[1] = *(const u32x4*)(qb + ko + 16);
        }
    }

    f32x8 o[4];
    float rmax[8], rsum[8];
    #pragma unroll
    for (int nt = 0; nt < 4; nt++)
        #pragma unroll
        for (int g = 0; g < 8; g++) o[nt][g] = 0.f;
    #pragma unroll
    for (int g = 0; g < 8; g++) { rmax[g] = -3.0e38f; rsum[g] = 0.f; }

    const int smax = t0 + 15;
    for (int s0 = 0; s0 <= smax; s0 += 32) {
        // ---- scores: two 16-col subtiles, K over H = 2 chunks of 32 ----
        f32x8 sc[2];
        #pragma unroll
        for (int sub = 0; sub < 2; sub++) {
            #pragma unroll
            for (int g = 0; g < 8; g++) sc[sub][g] = 0.f;
            const unsigned short* kb =
                qkv + (size_t)(b * kT + s0 + sub * 16 + l15) * (3 * kC) + kC + h * kH;
            FragB16 bk;
            #pragma unroll
            for (int cfr = 0; cfr < 2; cfr++) {
                int ko = cfr * 32 + half * 8;
                bk.q[0] = *(const u32x4*)(kb + ko);
                bk.q[1] = *(const u32x4*)(kb + ko + 16);
                sc[sub] = wmma_bf16(aq[cfr], bk, sc[sub]);
            }
        }
        // ---- scale + causal mask + online softmax ----
        const int row0 = t0 + half * 8;
        const int col0 = s0 + l15, col1 = s0 + 16 + l15;
        float alpha[8], p0[8], p1[8];
        #pragma unroll
        for (int g = 0; g < 8; g++) {
            const int row = row0 + g;
            float v0 = sc[0][g] * 0.125f; if (col0 > row) v0 = -3.0e38f;
            float v1 = sc[1][g] * 0.125f; if (col1 > row) v1 = -3.0e38f;
            float mx = fmaxf(v0, v1);
            #pragma unroll
            for (int m = 1; m < 16; m <<= 1) mx = fmaxf(mx, __shfl_xor(mx, m, 32));
            float nm = fmaxf(rmax[g], mx);
            float a  = __expf(rmax[g] - nm);
            rmax[g] = nm;
            v0 = __expf(v0 - nm);
            v1 = __expf(v1 - nm);
            float rs = v0 + v1;
            #pragma unroll
            for (int m = 1; m < 16; m <<= 1) rs += __shfl_xor(rs, m, 32);
            rsum[g] = rsum[g] * a + rs;
            alpha[g] = a; p0[g] = v0; p1[g] = v1;
        }
        #pragma unroll
        for (int nt = 0; nt < 4; nt++)
            #pragma unroll
            for (int g = 0; g < 8; g++) o[nt][g] *= alpha[g];

        // ---- repack P (C-layout f32) -> A-layout bf16 via per-wave LDS ----
        unsigned short* pb = pbuf[wave];
        #pragma unroll
        for (int g = 0; g < 8; g++) {
            const int r = half * 8 + g;
            pb[r * 40 + l15]      = f2bf(p0[g]);
            pb[r * 40 + 16 + l15] = f2bf(p1[g]);
        }
        FragB16 ap;
        {
            const unsigned short* p = pb + l15 * 40 + half * 8;
            ap.q[0] = *(const u32x4*)p;
            ap.q[1] = *(const u32x4*)(p + 16);
        }
        // ---- O += P(16x32) x V(32x64): 4 n-subtiles ----
        #pragma unroll
        for (int nt = 0; nt < 4; nt++) {
            FragB16 bv;
            const unsigned short* vb =
                vt + ((size_t)bh * kH + nt * 16 + l15) * kT + s0 + half * 8;
            bv.q[0] = *(const u32x4*)vb;
            bv.q[1] = *(const u32x4*)(vb + 16);
            o[nt] = wmma_bf16(ap, bv, o[nt]);
        }
    }

    // ---- normalize and store ctx[b,t, h*64 + n] ----
    #pragma unroll
    for (int g = 0; g < 8; g++) {
        const float inv = 1.0f / rsum[g];
        const int trow = t0 + half * 8 + g;
        #pragma unroll
        for (int nt = 0; nt < 4; nt++) {
            const int ncol = h * kH + nt * 16 + l15;
            ctx[(size_t)(b * kT + trow) * kC + ncol] = f2bf(o[nt][g] * inv);
        }
    }
}

// ---------------------------------------------------------------------------
// Launch
// ---------------------------------------------------------------------------
extern "C" void kernel_launch(void* const* d_in, const int* in_sizes, int n_in,
                              void* d_out, int out_size, void* d_ws, size_t ws_size,
                              hipStream_t stream) {
    (void)in_sizes; (void)n_in; (void)out_size; (void)ws_size;
    const float* x   = (const float*)d_in[0];
    const float* Wq  = (const float*)d_in[1];
    const float* Wk  = (const float*)d_in[2];
    const float* Wv  = (const float*)d_in[3];
    const float* Wo  = (const float*)d_in[4];
    const float* bo  = (const float*)d_in[5];
    const float* W1  = (const float*)d_in[6];
    const float* b1  = (const float*)d_in[7];
    const float* W2  = (const float*)d_in[8];
    const float* b2  = (const float*)d_in[9];
    const float* g1  = (const float*)d_in[10];
    const float* be1 = (const float*)d_in[11];
    const float* g2  = (const float*)d_in[12];
    const float* be2 = (const float*)d_in[13];
    float* out = (float*)d_out;

    char* ws = (char*)d_ws;
    // bf16 weights
    unsigned short* w_qkv = (unsigned short*)(ws + 0);                  //  6 MB
    unsigned short* w_o   = (unsigned short*)(ws + 6291456);            //  2 MB
    unsigned short* w_1   = (unsigned short*)(ws + 8388608);            //  6 MB
    unsigned short* w_2   = (unsigned short*)(ws + 14680064);           //  6 MB
    // activations
    unsigned short* xn    = (unsigned short*)(ws + 20971520);           //  8 MB (LN1 out; reused as LN2 out)
    unsigned short* qkv   = (unsigned short*)(ws + 29360128);           // 24 MB (reused as ffn1 out)
    unsigned short* vt    = (unsigned short*)(ws + 54525952);           //  8 MB
    unsigned short* ctx   = (unsigned short*)(ws + 62914560);           //  8 MB
    float*          x1    = (float*)(ws + 71303168);                    // 16 MB (x after attn residual)

    // 1) weight prep (bf16, [out,in] layouts)
    k_prep_qkv<<<(3 * kC * kC) / 256, 256, 0, stream>>>(Wq, Wk, Wv, w_qkv);
    k_cvt_bf16<<<(kC * kC) / 256, 256, 0, stream>>>(Wo, w_o, kC * kC);
    k_cvt_bf16<<<(kFF * kC) / 256, 256, 0, stream>>>(W1, w_1, kFF * kC);
    k_cvt_bf16<<<(kC * kFF) / 256, 256, 0, stream>>>(W2, w_2, kC * kFF);

    // 2) LN1
    k_layernorm<<<kM, 256, 0, stream>>>(x, g1, be1, xn);

    // 3) fused QKV projection: [4096,1024] x [3072,1024]^T -> bf16 [4096,3072]
    k_gemm<false, false, false, true><<<dim3(kFF / GBN, kM / GBM), 256, 0, stream>>>(
        xn, w_qkv, nullptr, nullptr, qkv, kM, kFF, kC);

    // 4) V transpose for PV fragments
    k_transpose_v<<<dim3(kT / 32, kB * kNH), 256, 0, stream>>>(qkv, vt);

    // 5) causal flash attention -> ctx bf16 [4096,1024]
    k_attention<<<dim3(kT / 128, kB * kNH), 256, 0, stream>>>(qkv, vt, ctx);

    // 6) output projection + bias + residual(x) -> x1 f32
    k_gemm<true, false, true, false><<<dim3(kC / GBN, kM / GBM), 256, 0, stream>>>(
        ctx, w_o, bo, x, x1, kM, kC, kC);

    // 7) LN2
    k_layernorm<<<kM, 256, 0, stream>>>(x1, g2, be2, xn);

    // 8) MLP up + ReLU -> bf16 [4096,3072] (reuse qkv buffer)
    k_gemm<true, true, false, true><<<dim3(kFF / GBN, kM / GBM), 256, 0, stream>>>(
        xn, w_1, b1, nullptr, qkv, kM, kFF, kC);

    // 9) MLP down + bias + residual(x1) -> d_out f32
    k_gemm<true, false, true, false><<<dim3(kC / GBN, kM / GBM), 256, 0, stream>>>(
        qkv, w_2, b2, x1, out, kM, kC, kFF);
}
